// MACE_30133490549677
// MI455X (gfx1250) — matrix operand order repaired and checked by reading, not compiled, not run on hardware
//
#include <hip/hip_runtime.h>
#include <math.h>

// ---------------- problem constants (match reference) ----------------
#define Nn   16384
#define Ee   131072
#define Ff   64
#define Kk   16
#define Zz   10
#define Gg   64
#define RMAXf 5.0f
#define INV_AVG 0.125f

typedef __attribute__((ext_vector_type(16))) __bf16 v16bf;
typedef __attribute__((ext_vector_type(8)))  float  v8f;

// float -> bf16 bits (round-to-nearest-even), stored as ushort to keep
// __bf16 out of host-visible signatures.
__device__ __forceinline__ unsigned short f2bfbits(float f) {
    unsigned u = __float_as_uint(f);
    unsigned r = (u + 0x7FFFu + ((u >> 16) & 1u)) >> 16;
    return (unsigned short)r;
}
__device__ __forceinline__ __bf16 bits2bf(unsigned short s) {
    __bf16 o; __builtin_memcpy(&o, &s, 2); return o;
}
// fast silu: v_exp_f32 + v_rcp_f32 (both TRANS, co-execute with WMMA);
// avoids the IEEE div_scale/div_fixup expansion of a plain divide.
__device__ __forceinline__ float siluf(float x) {
    return x * __builtin_amdgcn_rcpf(1.0f + __expf(-x));
}

// A-matrix (16x32 bf16) K index for vector element j, lane-half h:
// K = h*16 + j (contiguous, matches pack_b_kernel)
__device__ __forceinline__ int kofA(int j, int h) {
    return h * 16 + j;
}

__device__ __forceinline__ v16bf load_frag(const unsigned short* p) {
    return *(const v16bf*)p;
}
__device__ __forceinline__ v8f vzero8() {
    v8f c;
#pragma unroll
    for (int r = 0; r < 8; ++r) c[r] = 0.f;
    return c;
}
__device__ __forceinline__ v8f wmma_bf16(v16bf a, v16bf b, v8f c) {
    return __builtin_amdgcn_wmma_f32_16x16x32_bf16(false, a, false, b,
                                                   (short)0, c, false, false);
}

// ---------------- weight packing: row-major f32 [K,N] -> WMMA B frags ----
// frag layout: out[((kc*ntiles+nt)*32 + lane)*16 + j] = W[kc*32 + half*16 + j][nt*16 + lane%16]
__global__ void pack_b_kernel(const float* __restrict__ W, unsigned short* __restrict__ out,
                              int Ndim, int Kreal, int ntiles, int kchunks) {
    int idx = blockIdx.x * blockDim.x + threadIdx.x;
    int total = kchunks * ntiles * 512;
    if (idx >= total) return;
    int j    = idx & 15;
    int lane = (idx >> 4) & 31;
    int u    = idx >> 9;
    int nt   = u % ntiles;
    int kc   = u / ntiles;
    int half = lane >> 4;
    int n    = nt * 16 + (lane & 15);
    int k    = kc * 32 + half * 16 + j;
    float v  = (k < Kreal) ? W[k * Ndim + n] : 0.0f;
    out[idx] = f2bfbits(v);
}

// ---------------- per-edge geometry: Y [E,16] and radial basis [E,8] -----
__global__ void geom_kernel(const float* __restrict__ pos,
                            const int* __restrict__ src, const int* __restrict__ dst,
                            float* __restrict__ Yout, float* __restrict__ efout) {
    int e = blockIdx.x * blockDim.x + threadIdx.x;
    if (e >= Ee) return;
    int s = src[e], d = dst[e];
    float vx = pos[d*3+0] - pos[s*3+0];
    float vy = pos[d*3+1] - pos[s*3+1];
    float vz = pos[d*3+2] - pos[s*3+2];
    float len = sqrtf(vx*vx + vy*vy + vz*vz + 1e-12f);
    float inv = 1.0f / len;
    float x = vx*inv, y = vy*inv, z = vz*inv;
    float xx = x*x, yy = y*y, zz = z*z;
    const float s3 = 1.7320508075688772f, s5 = 2.2360679774997896f;
    const float s7 = 2.6457513110645907f, s15 = 3.8729833462074170f;
    const float s21 = 4.5825756949558400f, s35 = 5.9160797830996160f;
    const float s105 = 10.246950765959598f;
    float Yv[16];
    Yv[0]  = 1.0f;
    Yv[1]  = s3 * x;  Yv[2] = s3 * y;  Yv[3] = s3 * z;
    Yv[4]  = s15 * x * y;
    Yv[5]  = s15 * y * z;
    Yv[6]  = 0.5f * s5 * (3.0f * zz - 1.0f);
    Yv[7]  = s15 * x * z;
    Yv[8]  = 0.5f * s15 * (xx - yy);
    Yv[9]  = 0.25f * s35 * y * (3.0f * xx - yy);
    Yv[10] = 0.5f  * s105 * x * y * z;
    Yv[11] = 0.25f * s21 * y * (5.0f * zz - 1.0f);
    Yv[12] = 0.5f  * s7  * z * (5.0f * zz - 3.0f);
    Yv[13] = 0.25f * s21 * x * (5.0f * zz - 1.0f);
    Yv[14] = 0.25f * s105 * z * (xx - yy);
    Yv[15] = 0.25f * s35 * x * (xx - 3.0f * yy);
#pragma unroll
    for (int k = 0; k < 16; ++k) Yout[(size_t)e*16 + k] = Yv[k];

    float uu = len / RMAXf;
    float safe = fmaxf(len, 1e-6f);
    float u5 = uu*uu*uu*uu*uu;
    float env = 1.0f - 21.0f*u5 + 35.0f*u5*uu - 15.0f*u5*uu*uu;   // p=5 cutoff
    env = (uu < 1.0f) ? env : 0.0f;
    const float pref = 0.63245553203367580f;  // sqrt(2/RMAX)
    const float pi = 3.14159265358979323846f;
    float scale = pref / safe * env;
#pragma unroll
    for (int nb = 1; nb <= 8; ++nb)
        efout[(size_t)e*8 + nb - 1] = scale * sinf((float)nb * pi * uu);
}

// ---------------- init: h[:,0,:] = W_embed[species], node_es = E0, out=0 --
__global__ void init_kernel(const float* __restrict__ W_embed, const float* __restrict__ aE,
                            const int* __restrict__ species, float* __restrict__ h,
                            float* __restrict__ node_es, float* __restrict__ out) {
    int idx = blockIdx.x * blockDim.x + threadIdx.x;   // N*1024 threads
    int n = idx >> 10;
    int k = (idx >> 6) & 15;
    int f = idx & 63;
    h[idx] = (k == 0) ? W_embed[species[n] * Ff + f] : 0.0f;
    if (idx < Nn) node_es[idx] = aE[species[idx]];
    if (idx < Gg) out[idx] = 0.0f;
}

__global__ void zero_kernel(float* __restrict__ p, int count) {
    int i = blockIdx.x * blockDim.x + threadIdx.x;
    if (i < count) p[i] = 0.0f;
}

// ---------------- fused edge kernel: radial MLP (WMMA) + TP + scatter ----
// one wave = 16 edges; 8 waves/block
__global__ __launch_bounds__(256) void edge_kernel(
    const float* __restrict__ ef, const float* __restrict__ Y,
    const int* __restrict__ src, const int* __restrict__ dst,
    const float* __restrict__ h,
    const unsigned short* __restrict__ pW1, const unsigned short* __restrict__ pW2,
    const unsigned short* __restrict__ pW3, const unsigned short* __restrict__ pWout,
    float* __restrict__ Acc) {
    __shared__ unsigned short Hbuf[8][16][64];   // activations (bf16 bits)
    __shared__ float hsrcS[8][16][64];
    __shared__ float Ys[8][16][16];
    __shared__ int   dstS[8][16];

    int wave = threadIdx.x >> 5;
    int lane = threadIdx.x & 31;
    int row  = lane & 15;
    int half = lane >> 4;
    int e_row = (blockIdx.x * 8 + wave) * 16 + row;

    if (half == 0) dstS[wave][row] = dst[e_row];
#pragma unroll
    for (int kk = 0; kk < 8; ++kk)
        Ys[wave][row][half*8 + kk] = Y[(size_t)e_row*16 + half*8 + kk];
    {
        int s = src[e_row];
#pragma unroll
        for (int ff = 0; ff < 32; ++ff)
            hsrcS[wave][row][half*32 + ff] = h[(size_t)s*1024 + half*32 + ff];
    }
    __syncthreads();

    // ---- stage 1: [16,8(pad 32)] @ [32,64] -> silu ----
    v16bf a1;
#pragma unroll
    for (int j = 0; j < 16; ++j) a1[j] = bits2bf(0);
    if (half == 0) {
#pragma unroll
        for (int j = 0; j < 8; ++j) a1[j] = bits2bf(f2bfbits(ef[(size_t)e_row*8 + j]));
    }
    {
        v8f c[4];
#pragma unroll
        for (int nt = 0; nt < 4; ++nt) {
            v16bf b = load_frag(pW1 + ((size_t)(nt*32 + lane)) * 16);
            c[nt] = wmma_bf16(a1, b, vzero8());
        }
#pragma unroll
        for (int nt = 0; nt < 4; ++nt)
#pragma unroll
            for (int r = 0; r < 8; ++r)
                Hbuf[wave][half*8 + r][nt*16 + row] = f2bfbits(siluf(c[nt][r]));
    }
    __syncthreads();

    // ---- stages 2,3: [16,64] @ [64,64] -> silu ----
    for (int stage = 0; stage < 2; ++stage) {
        const unsigned short* pw = (stage == 0) ? pW2 : pW3;
        v16bf a[2];
#pragma unroll
        for (int kc = 0; kc < 2; ++kc)
#pragma unroll
            for (int j = 0; j < 16; ++j)
                a[kc][j] = bits2bf(Hbuf[wave][row][kc*32 + kofA(j, half)]);
        __syncthreads();   // reads done before overwrite
        v8f c[4];
#pragma unroll
        for (int nt = 0; nt < 4; ++nt) c[nt] = vzero8();
#pragma unroll
        for (int kc = 0; kc < 2; ++kc)
#pragma unroll
            for (int nt = 0; nt < 4; ++nt) {
                v16bf b = load_frag(pw + ((size_t)((kc*4 + nt)*32 + lane)) * 16);
                c[nt] = wmma_bf16(a[kc], b, c[nt]);
            }
#pragma unroll
        for (int nt = 0; nt < 4; ++nt)
#pragma unroll
            for (int r = 0; r < 8; ++r)
                Hbuf[wave][half*8 + r][nt*16 + row] = f2bfbits(siluf(c[nt][r]));
        __syncthreads();
    }

    // ---- stage 4: [16,64] @ [64,1024]; fuse R*Y*h_src/AVG scatter ----
    v16bf aH[2];
#pragma unroll
    for (int kc = 0; kc < 2; ++kc)
#pragma unroll
        for (int j = 0; j < 16; ++j)
            aH[kc][j] = bits2bf(Hbuf[wave][row][kc*32 + kofA(j, half)]);

    // register-cache per-lane reuse data: h_src/AVG (4 cols x 8 rows), dst nodes
    float hs[4][8];
#pragma unroll
    for (int ft = 0; ft < 4; ++ft)
#pragma unroll
        for (int r = 0; r < 8; ++r)
            hs[ft][r] = hsrcS[wave][half*8 + r][ft*16 + row] * INV_AVG;
    int dnode[8];
#pragma unroll
    for (int r = 0; r < 8; ++r) dnode[r] = dstS[wave][half*8 + r];

    for (int kidx = 0; kidx < 16; ++kidx) {        // spherical harmonic component
        float ys[8];
#pragma unroll
        for (int r = 0; r < 8; ++r) ys[r] = Ys[wave][half*8 + r][kidx];
#pragma unroll
        for (int ft = 0; ft < 4; ++ft) {           // 16-wide channel tile
            int nt = kidx * 4 + ft;
            // prefetch next tile's fragments into cache (global_prefetch_b8)
            if (nt + 1 < 64) {
                __builtin_prefetch(pWout + ((size_t)((nt + 1) * 32 + lane)) * 16, 0, 3);
                __builtin_prefetch(pWout + ((size_t)((64 + nt + 1) * 32 + lane)) * 16, 0, 3);
            }
            v8f c = vzero8();
#pragma unroll
            for (int kc = 0; kc < 2; ++kc) {
                v16bf b = load_frag(pWout + ((size_t)((kc*64 + nt)*32 + lane)) * 16);
                c = wmma_bf16(aH[kc], b, c);
            }
            int f = ft*16 + row;                   // channel
#pragma unroll
            for (int r = 0; r < 8; ++r) {
                float val = c[r] * ys[r] * hs[ft][r];
                atomicAdd(&Acc[(size_t)dnode[r]*1024 + kidx*64 + f], val);
            }
        }
    }
}

// ---------------- node kernel: sc (WMMA) + invariants + readout ----------
// one wave = 1 node; 8 waves/block
__global__ __launch_bounds__(256) void node_kernel(
    const float* __restrict__ Acc, float* __restrict__ h,
    const int* __restrict__ species, const unsigned short* __restrict__ pWsc,
    const float* __restrict__ pw, const float* __restrict__ read1_w,
    const float* __restrict__ read2_Wa, const float* __restrict__ read2_wb,
    float* __restrict__ node_es, const int* __restrict__ batch,
    float* __restrict__ out, int layer) {
    __shared__ float h0row[8][64];
    int wave = threadIdx.x >> 5;
    int lane = threadIdx.x & 31;
    int row  = lane & 15;
    int half = lane >> 4;
    int n = blockIdx.x * 8 + wave;
    int z = species[n];

    // sc = h[n] (16x64) @ Wsc[layer][z] (64x64), A rows = k, K = channel c
    v16bf a[2];
#pragma unroll
    for (int kc = 0; kc < 2; ++kc)
#pragma unroll
        for (int j = 0; j < 16; ++j)
            a[kc][j] = bits2bf(f2bfbits(h[(size_t)n*1024 + row*64 + kc*32 + kofA(j, half)]));
    const unsigned short* wz = pWsc + (size_t)((layer*Zz + z) * 8) * 512;
    v8f sc[4];
#pragma unroll
    for (int nt = 0; nt < 4; ++nt) sc[nt] = vzero8();
#pragma unroll
    for (int kc = 0; kc < 2; ++kc)
#pragma unroll
        for (int nt = 0; nt < 4; ++nt) {
            v16bf b = load_frag(wz + ((size_t)((kc*4 + nt)*32 + lane)) * 16);
            sc[nt] = wmma_bf16(a[kc], b, sc[nt]);
        }

    // elementwise: h = A*pw0 (+ row0: inv2*pw1 + inv3*pw2) + sc
#pragma unroll
    for (int nt = 0; nt < 4; ++nt) {
        int fcol = nt*16 + row;
        float p0 = pw[(layer*3 + 0)*64 + fcol];
        float p1 = pw[(layer*3 + 1)*64 + fcol];
        float p2 = pw[(layer*3 + 2)*64 + fcol];
        float av[8]; float ss = 0.f;
#pragma unroll
        for (int r = 0; r < 8; ++r) {
            av[r] = Acc[(size_t)n*1024 + (half*8 + r)*64 + fcol];
            ss += av[r]*av[r];
        }
        float inv2 = ss + __shfl_xor(ss, 16);          // sum both row-halves of column
        float a0   = __shfl(av[0], row);               // A[0,f] lives in half0 lane
        float inv3 = inv2 * a0;
#pragma unroll
        for (int r = 0; r < 8; ++r) {
            int m = half*8 + r;
            float hv = av[r]*p0 + sc[nt][r];
            if (m == 0) hv += inv2*p1 + inv3*p2;
            h[(size_t)n*1024 + m*64 + fcol] = hv;
            if (m == 0) h0row[wave][fcol] = hv;
        }
    }
    __syncthreads();

    if (layer == 0) {
        float p = 0.f;
        for (int f = lane; f < 64; f += 32) p += h0row[wave][f] * read1_w[f];
#pragma unroll
        for (int o = 16; o >= 1; o >>= 1) p += __shfl_xor(p, o);
        if (lane == 0) node_es[n] += p;
    } else {
        float acc = 0.f;
        if (half == 0) {
            float t = 0.f;
#pragma unroll
            for (int f = 0; f < 64; ++f) t += h0row[wave][f] * read2_Wa[f*16 + row];
            acc = siluf(t) * read2_wb[row];
        }
#pragma unroll
        for (int o = 16; o >= 1; o >>= 1) acc += __shfl_xor(acc, o);
        if (lane == 0) atomicAdd(&out[batch[n]], node_es[n] + acc);
    }
}

// ---------------------------- host launcher ------------------------------
extern "C" void kernel_launch(void* const* d_in, const int* in_sizes, int n_in,
                              void* d_out, int out_size, void* d_ws, size_t ws_size,
                              hipStream_t stream) {
    (void)in_sizes; (void)n_in; (void)out_size; (void)ws_size;
    const float* positions  = (const float*)d_in[0];
    const float* aE         = (const float*)d_in[1];
    const float* W_embed    = (const float*)d_in[2];
    const float* rW1        = (const float*)d_in[3];
    const float* rW2        = (const float*)d_in[4];
    const float* rW3        = (const float*)d_in[5];
    const float* rWout      = (const float*)d_in[6];
    const float* Wsc        = (const float*)d_in[7];
    const float* pw         = (const float*)d_in[8];
    const float* read1_w    = (const float*)d_in[9];
    const float* read2_Wa   = (const float*)d_in[10];
    const float* read2_wb   = (const float*)d_in[11];
    const int*   species    = (const int*)d_in[12];
    const int*   edge_index = (const int*)d_in[13];
    const int*   batch      = (const int*)d_in[14];
    const int* src = edge_index;
    const int* dst = edge_index + Ee;
    float* out = (float*)d_out;

    // workspace layout (~147.6 MB)
    size_t off = 0;
    char* base = (char*)d_ws;
    auto carve = [&](size_t bytes) { char* p = base + off; off += (bytes + 255) & ~(size_t)255; return p; };
    float* Yw   = (float*)carve((size_t)Ee * 16 * 4);
    float* efw  = (float*)carve((size_t)Ee * 8 * 4);
    float* hw   = (float*)carve((size_t)Nn * 1024 * 4);
    float* Accw = (float*)carve((size_t)Nn * 1024 * 4);
    float* nesw = (float*)carve((size_t)Nn * 4);
    unsigned short* pW1   = (unsigned short*)carve((size_t)2 * 4   * 512 * 2);
    unsigned short* pW2   = (unsigned short*)carve((size_t)2 * 8   * 512 * 2);
    unsigned short* pW3   = (unsigned short*)carve((size_t)2 * 8   * 512 * 2);
    unsigned short* pWout = (unsigned short*)carve((size_t)2 * 128 * 512 * 2);
    unsigned short* pWsc  = (unsigned short*)carve((size_t)2 * Zz * 8 * 512 * 2);

    // pack weights into WMMA B-fragment order (bf16)
    for (int l = 0; l < 2; ++l) {
        pack_b_kernel<<<8,   256, 0, stream>>>(rW1 + (size_t)l*8*64,   pW1 + (size_t)l*4*512,   64,   8,  4, 1);
        pack_b_kernel<<<16,  256, 0, stream>>>(rW2 + (size_t)l*64*64,  pW2 + (size_t)l*8*512,   64,   64, 4, 2);
        pack_b_kernel<<<16,  256, 0, stream>>>(rW3 + (size_t)l*64*64,  pW3 + (size_t)l*8*512,   64,   64, 4, 2);
        pack_b_kernel<<<256, 256, 0, stream>>>(rWout + (size_t)l*64*1024, pWout + (size_t)l*128*512, 1024, 64, 64, 2);
        for (int z = 0; z < Zz; ++z)
            pack_b_kernel<<<16, 256, 0, stream>>>(Wsc + ((size_t)l*Zz + z)*4096,
                                                  pWsc + ((size_t)(l*Zz + z))*8*512, 64, 64, 4, 2);
    }

    geom_kernel<<<Ee/256, 256, 0, stream>>>(positions, src, dst, Yw, efw);
    init_kernel<<<(Nn*1024)/256, 256, 0, stream>>>(W_embed, aE, species, hw, nesw, out);

    for (int l = 0; l < 2; ++l) {
        zero_kernel<<<(Nn*1024)/256, 256, 0, stream>>>(Accw, Nn*1024);
        edge_kernel<<<Ee/128, 256, 0, stream>>>(efw, Yw, src, dst, hw,
                                                pW1 + (size_t)l*4*512,
                                                pW2 + (size_t)l*8*512,
                                                pW3 + (size_t)l*8*512,
                                                pWout + (size_t)l*128*512,
                                                Accw);
        node_kernel<<<Nn/8, 256, 0, stream>>>(Accw, hw, species, pWsc, pw,
                                              read1_w, read2_Wa, read2_wb,
                                              nesw, batch, out, l);
    }
}
